// DIN_78374563217689
// MI455X (gfx1250) — matrix operand order repaired and checked by reading, not compiled, not run on hardware
//
#include <hip/hip_runtime.h>

#define DIN_EPS 1e-5f

typedef __attribute__((ext_vector_type(16))) _Float16 v16h;
typedef __attribute__((ext_vector_type(8)))  float    v8f;

// Input pointer bundle (pytree flatten order: dict keys sorted; layer dicts
// flatten as W, alpha, b, bn_b, bn_g, bn_m, bn_v, d_m, d_v).
struct DinParams {
  const float* x;      // (total, 16) f32
  const int*   off;    // (B*5,) i32
  // attn MLP: 128 -> 16 -> 8 -> 1
  const float *aWo, *abo;                                                     // (8,1),(1,)
  const float *aW1,*a_al1,*ab1,*a_bnb1,*a_bng1,*a_bnm1,*a_bnv1,*a_dm1,*a_dv1; // layer0 (128,16)
  const float *aW2,*a_al2,*ab2,*a_bnb2,*a_bng2,*a_bnm2,*a_bnv2,*a_dm2,*a_dv2; // layer1 (16,8)
  // final MLP: 80 -> 32 -> 16 -> 1 (+sigmoid)
  const float *mWo, *mbo;                                                     // (16,1),(1,)
  const float *mW1,*m_al1,*mb1,*m_bnb1,*m_bng1,*m_bnm1,*m_bnv1,*m_dm1,*m_dv1; // layer0 (80,32)
  const float *mW2,*m_al2,*mb2,*m_bnb2,*m_bng2,*m_bnm2,*m_bnv2,*m_dm2,*m_dv2; // layer1 (32,16)
  float* out;          // (B,) f32
};

__device__ __forceinline__ float sigmoidf(float z) { return 1.f / (1.f + __expf(-z)); }

// Wave-private LDS handoff: DS ops from one wave execute in order; drain them
// and pin compiler ordering so cross-lane LDS data written by this wave is
// readable by its own lanes without a workgroup barrier.
__device__ __forceinline__ void wave_ds_fence() {
  __builtin_amdgcn_wave_barrier();
  asm volatile("s_wait_dscnt 0x0" ::: "memory");
  __builtin_amdgcn_wave_barrier();
}

// All of this wave's outstanding async-to-LDS copies have landed.
__device__ __forceinline__ void wave_async_wait() {
  __builtin_amdgcn_wave_barrier();
  asm volatile("s_wait_asynccnt 0x0" ::: "memory");
  __builtin_amdgcn_wave_barrier();
}

// Per-lane 64B global->LDS async copy (CDNA5 ASYNCcnt path, bypasses VGPRs).
// INST_OFFSET is added to BOTH the memory and LDS addresses (ISA 10.x async),
// so one base pair serves all four b128 beats.
__device__ __forceinline__ void async_copy64B_to_lds(unsigned lds_byte_addr,
                                                     const float* gsrc) {
  asm volatile(
      "global_load_async_to_lds_b128 %0, %1, off\n\t"
      "global_load_async_to_lds_b128 %0, %1, off offset:16\n\t"
      "global_load_async_to_lds_b128 %0, %1, off offset:32\n\t"
      "global_load_async_to_lds_b128 %0, %1, off offset:48"
      :: "v"(lds_byte_addr), "v"(gsrc) : "memory");
}

__device__ __forceinline__ unsigned lds_addr_of(const void* p) {
  return (unsigned)(unsigned long long)p;  // LDS aperture: addr[31:0] = LDS offset
}

// One block per batch element b. 128 threads = 4 wave32s; wave w owns sequence
// tiles {w, w+4, w+8, w+12} of 16 timesteps each (T=200 -> 13 tiles max).
// All tile-loop LDS traffic is wave-private: no block barriers in the loop.
__launch_bounds__(128, 1)
__global__ void din_fused_kernel(DinParams P) {
  const int b   = blockIdx.x;
  const int tid = threadIdx.x;
  const int w   = tid >> 5;   // wave id 0..3
  const int ln  = tid & 31;   // lane 0..31
  const int m   = ln & 15;    // M-row (A) / N-col (B,C) index
  const int hf  = ln >> 4;    // lane half

  __shared__ __align__(16) float s_seq[4][2][16 * 32]; // per-wave double-buffered seq tile
  __shared__ __align__(16) float s_h1[4][16 * 16];     // per-wave layer-1 activations (t x 16)
  __shared__ float s_tgt[32];                          // target features (bcast over t)
  __shared__ float s_score[4][16];                     // per-wave attention scores
  __shared__ float s_pool4[4][32];                     // per-wave partial pooled
  __shared__ float s_emb[80];                          // [user16 | pooled32 | target32]
  __shared__ float s_l2c[8][7];                        // attn layer2 fused consts + Wout
  __shared__ float s_aW2[128];                         // attn layer2 weights (16x8)
  __shared__ float s_m1[32], s_m2[16];                 // final-MLP intermediates

  const long o0 = P.off[b * 5 + 0];
  const long o1 = P.off[b * 5 + 1];
  const long o2 = P.off[b * 5 + 2];
  const long o3 = P.off[b * 5 + 3];
  const long o4 = P.off[b * 5 + 4];
  const int len = (int)(o2 - o1);

  // ---- init: target, user, layer2 constants (block-shared, one barrier) ----
  if (tid < 32)
    s_tgt[tid] = (tid < 16) ? P.x[o2 * 16 + tid] : P.x[o4 * 16 + (tid - 16)];
  if (tid >= 64 && tid < 80)
    s_emb[tid - 64] = P.x[o0 * 16 + (tid - 64)];           // user
  if (tid >= 32 && tid < 40) {                             // attn layer2 consts
    const int o = tid - 32;
    const float sc = P.a_bng2[o] * rsqrtf(P.a_bnv2[o] + DIN_EPS);
    s_l2c[o][0] = P.ab2[o];
    s_l2c[o][1] = sc;
    s_l2c[o][2] = P.a_bnb2[o] - P.a_bnm2[o] * sc;
    s_l2c[o][3] = rsqrtf(P.a_dv2[o] + DIN_EPS);
    s_l2c[o][4] = -P.a_dm2[o] * s_l2c[o][3];
    s_l2c[o][5] = P.a_al2[o];
    s_l2c[o][6] = P.aWo[o];
  }
  s_aW2[tid] = P.aW2[tid];                                 // 128 = blockDim
  __syncthreads();

  // ---- per-wave register preload ----
  // A-frag (16-bit 16x32) lane layout: row M = ln&15; element j holds
  // K = hf*8 + j (j<8) or 16 + hf*8 + (j-8) (j>=8).   [ISA 7.12.2]
  float ltg[16];
  v16h a0;  // tgt section of din: identical for every tile -> hoisted
#pragma unroll
  for (int j = 0; j < 16; ++j) {
    const int c = (j < 8) ? (hf * 8 + j) : (16 + hf * 8 + (j - 8));
    ltg[j] = s_tgt[c];
    a0[j]  = (_Float16)ltg[j];
  }
  // B-frags (32x16 per fragment): col N = ln&15; element j -> K = hf*8 + j (j<8) or 16 + hf*8 + (j-8) (j>=8).
  v16h bw0, bw1, bw2, bw3;
#pragma unroll
  for (int j = 0; j < 16; ++j) {
    const int kb = (j < 8) ? (hf * 8 + j) : (16 + hf * 8 + (j - 8));
    bw0[j] = (_Float16)P.aW1[(0 * 32 + kb) * 16 + m];
    bw1[j] = (_Float16)P.aW1[(1 * 32 + kb) * 16 + m];
    bw2[j] = (_Float16)P.aW1[(2 * 32 + kb) * 16 + m];
    bw3[j] = (_Float16)P.aW1[(3 * 32 + kb) * 16 + m];
  }
  // Fused layer-1 epilogue constants for hidden unit n = m.
  const float bias1 = P.ab1[m];
  const float bns = P.a_bng1[m] * rsqrtf(P.a_bnv1[m] + DIN_EPS);
  const float bnh = P.a_bnb1[m] - P.a_bnm1[m] * bns;
  const float dsc = rsqrtf(P.a_dv1[m] + DIN_EPS);
  const float dsh = -P.a_dm1[m] * dsc;
  const float al1 = P.a_al1[m];

  float pv = 0.f;  // this lane's pooled[d=ln] partial

  // ---- prologue: async-stage this wave's first seq tile into buffer 0 ----
  // Lane copies row t = t0 + m, half hf (16 contiguous floats = 64 B).
  if (w * 16 < len) {
    int t = w * 16 + m;
    if (t > len - 1) t = len - 1;  // clamp (masked rows: value irrelevant)
    async_copy64B_to_lds(lds_addr_of(&s_seq[w][0][m * 32 + hf * 16]),
                         P.x + ((hf ? o3 : o1) + t) * 16);
  }

  // ---- tile loop: fully wave-local (no block barriers) ----
  for (int it = 0; it < 4; ++it) {
    const int t0 = (w + 4 * it) * 16;
    const bool active = (t0 < len);  // wave-uniform
    const int buf = it & 1;

    if (active) wave_async_wait();   // current tile's seq has landed in LDS
    wave_ds_fence();                 // prior LDS reads of buf^1 have drained
                                     // (async writes are NOT ordered with DS)
    const int nt0 = t0 + 64;         // next tile for this wave
    if (nt0 < len) {                 // prefetch it into the other buffer:
      int t = nt0 + m;               // overlaps the whole compute below
      if (t > len - 1) t = len - 1;
      async_copy64B_to_lds(lds_addr_of(&s_seq[w][buf ^ 1][m * 32 + hf * 16]),
                           P.x + ((hf ? o3 : o1) + t) * 16);
    }

    // Phase B: build A-frags, 4x WMMA (K=128), fused bias+BN+Dice epilogue.
    if (active) {
      float ls[16];
      const float* sr = &s_seq[w][buf][m * 32];
#pragma unroll
      for (int j = 0; j < 16; ++j) {
        const int c = (j < 8) ? (hf * 8 + j) : (16 + hf * 8 + (j - 8));
        ls[j] = sr[c];
      }
      v16h a1, a2, a3;
#pragma unroll
      for (int j = 0; j < 16; ++j) {
        a1[j] = (_Float16)ls[j];
        a2[j] = (_Float16)(ltg[j] - ls[j]);
        a3[j] = (_Float16)(ltg[j] * ls[j]);
      }
      v8f acc;
#pragma unroll
      for (int v = 0; v < 8; ++v) acc[v] = bias1;  // bias folded into C
      acc = __builtin_amdgcn_wmma_f32_16x16x32_f16(false, a0, false, bw0, (short)0, acc, false, false);
      acc = __builtin_amdgcn_wmma_f32_16x16x32_f16(false, a1, false, bw1, (short)0, acc, false, false);
      acc = __builtin_amdgcn_wmma_f32_16x16x32_f16(false, a2, false, bw2, (short)0, acc, false, false);
      acc = __builtin_amdgcn_wmma_f32_16x16x32_f16(false, a3, false, bw3, (short)0, acc, false, false);
#pragma unroll
      for (int v = 0; v < 8; ++v) {  // C layout: M = v + 8*hf, N = m
        const float y = acc[v] * bns + bnh;            // BN
        const float p = sigmoidf(y * dsc + dsh);       // Dice gate
        s_h1[w][(v + 8 * hf) * 16 + m] = y * (al1 + p * (1.f - al1));
      }
    }
    wave_ds_fence();

    // Phase C: 16->8->1 tail per timestep (lane = timestep), masked scores.
    if (active && ln < 16) {
      float score = 0.f;
      if (t0 + m < len) {
        float hh[16];
#pragma unroll
        for (int q = 0; q < 4; ++q) {
          const float4 t4 = ((const float4*)&s_h1[w][m * 16])[q];
          hh[q * 4 + 0] = t4.x; hh[q * 4 + 1] = t4.y;
          hh[q * 4 + 2] = t4.z; hh[q * 4 + 3] = t4.w;
        }
        float sacc = P.abo[0];
#pragma unroll
        for (int o = 0; o < 8; ++o) {
          float z = s_l2c[o][0];
#pragma unroll
          for (int n = 0; n < 16; ++n) z += hh[n] * s_aW2[n * 8 + o];
          const float y = z * s_l2c[o][1] + s_l2c[o][2];
          const float p = sigmoidf(y * s_l2c[o][3] + s_l2c[o][4]);
          const float a = s_l2c[o][5];
          sacc += y * (a + p * (1.f - a)) * s_l2c[o][6];
        }
        score = sacc;
      }
      s_score[w][m] = score;
    }
    wave_ds_fence();

    // Phase D: pooled[d=ln] += sum_t score[t] * seq[t][d].
    if (active) {
      const float* sq = &s_seq[w][buf][0];
      const float* sc = &s_score[w][0];
#pragma unroll
      for (int r = 0; r < 16; ++r) pv += sc[r] * sq[r * 32 + ln];
    }
  }

  // ---- deterministic cross-wave pooled reduction ----
  s_pool4[w][ln] = pv;
  __syncthreads();
  if (tid < 32) {
    s_emb[16 + tid] = s_pool4[0][tid] + s_pool4[1][tid] +
                      s_pool4[2][tid] + s_pool4[3][tid];
    s_emb[48 + tid] = s_tgt[tid];
  }
  __syncthreads();

  // ---- final MLP 80 -> 32 -> 16 -> 1 + sigmoid ----
  if (tid < 32) {
    const int n = tid;
    float z = P.mb1[n];
    for (int k = 0; k < 80; ++k) z += s_emb[k] * P.mW1[k * 32 + n];
    const float sc = P.m_bng1[n] * rsqrtf(P.m_bnv1[n] + DIN_EPS);
    const float y  = z * sc + (P.m_bnb1[n] - P.m_bnm1[n] * sc);
    const float p  = sigmoidf((y - P.m_dm1[n]) * rsqrtf(P.m_dv1[n] + DIN_EPS));
    const float a  = P.m_al1[n];
    s_m1[n] = y * (a + p * (1.f - a));
  }
  __syncthreads();
  if (tid < 16) {
    const int n = tid;
    float z = P.mb2[n];
    for (int k = 0; k < 32; ++k) z += s_m1[k] * P.mW2[k * 16 + n];
    const float sc = P.m_bng2[n] * rsqrtf(P.m_bnv2[n] + DIN_EPS);
    const float y  = z * sc + (P.m_bnb2[n] - P.m_bnm2[n] * sc);
    const float p  = sigmoidf((y - P.m_dm2[n]) * rsqrtf(P.m_dv2[n] + DIN_EPS));
    const float a  = P.m_al2[n];
    s_m2[n] = y * (a + p * (1.f - a));
  }
  __syncthreads();
  if (tid == 0) {
    float z = P.mbo[0];
    for (int k = 0; k < 16; ++k) z += s_m2[k] * P.mWo[k];
    P.out[b] = sigmoidf(z);
  }
}

extern "C" void kernel_launch(void* const* d_in, const int* in_sizes, int n_in,
                              void* d_out, int out_size, void* d_ws, size_t ws_size,
                              hipStream_t stream) {
  (void)in_sizes; (void)n_in; (void)d_ws; (void)ws_size;
  DinParams P;
  const float** fp = (const float**)d_in;
  P.x      = fp[0];
  P.off    = (const int*)d_in[1];
  // attn: W_out, b_out, layers[0]{W,alpha,b,bn_b,bn_g,bn_m,bn_v,d_m,d_v}, layers[1]{...}
  P.aWo    = fp[2];  P.abo    = fp[3];
  P.aW1    = fp[4];  P.a_al1  = fp[5];  P.ab1    = fp[6];
  P.a_bnb1 = fp[7];  P.a_bng1 = fp[8];  P.a_bnm1 = fp[9];
  P.a_bnv1 = fp[10]; P.a_dm1  = fp[11]; P.a_dv1  = fp[12];
  P.aW2    = fp[13]; P.a_al2  = fp[14]; P.ab2    = fp[15];
  P.a_bnb2 = fp[16]; P.a_bng2 = fp[17]; P.a_bnm2 = fp[18];
  P.a_bnv2 = fp[19]; P.a_dm2  = fp[20]; P.a_dv2  = fp[21];
  // mlp: same ordering
  P.mWo    = fp[22]; P.mbo    = fp[23];
  P.mW1    = fp[24]; P.m_al1  = fp[25]; P.mb1    = fp[26];
  P.m_bnb1 = fp[27]; P.m_bng1 = fp[28]; P.m_bnm1 = fp[29];
  P.m_bnv1 = fp[30]; P.m_dm1  = fp[31]; P.m_dv1  = fp[32];
  P.mW2    = fp[33]; P.m_al2  = fp[34]; P.mb2    = fp[35];
  P.m_bnb2 = fp[36]; P.m_bng2 = fp[37]; P.m_bnm2 = fp[38];
  P.m_bnv2 = fp[39]; P.m_dm2  = fp[40]; P.m_dv2  = fp[41];
  P.out    = (float*)d_out;

  din_fused_kernel<<<out_size, 128, 0, stream>>>(P);
}
